// UVNetGraphModel_1769526526584
// MI455X (gfx1250) — matrix operand order repaired and checked by reading, not compiled, not run on hardware
//
#include <hip/hip_runtime.h>
#include <math.h>

// ---------------------------------------------------------------------------
// UVNet-style GCN forward for MI455X (gfx1250, wave32, WMMA).
//   sparse graph propagation : COO scatter w/ global f32 atomics
//   dense GEMMs              : v_wmma_f32_16x16x32_f16 (f16 in, f32 acc),
//                              padded so fragment loads are unguarded (A loads b128)
// ---------------------------------------------------------------------------

typedef __attribute__((ext_vector_type(16))) _Float16 v16h;
typedef __attribute__((ext_vector_type(8)))  float    v8f;

#define GN 12288
#define GE 393216
#define GB 16
#define GNC 11
#define BN_EPS 1e-5f

// ---------------------------- small utility kernels ------------------------

__global__ void k_init_xpad_deg(const float* __restrict__ x,
                                float* __restrict__ xpad,
                                float* __restrict__ deg) {
    int i = blockIdx.x * blockDim.x + threadIdx.x;
    if (i >= GN * 16) return;
    int node = i >> 4, f = i & 15;
    xpad[i] = (f < 15) ? x[node * 15 + f] : 0.f;   // pad F_IN=15 -> 16
    if (f == 0) deg[node] = 1.f;                   // self-loop weight
}

__global__ void k_edge_weight_deg(const float* __restrict__ ea,
                                  const float* __restrict__ w_edge,
                                  const float* __restrict__ b_edge,
                                  const int* __restrict__ row,
                                  float* __restrict__ ew,
                                  float* __restrict__ deg) {
    int e = blockIdx.x * blockDim.x + threadIdx.x;
    if (e >= GE) return;
    float z = ea[2 * e] * w_edge[0] + ea[2 * e + 1] * w_edge[1] + b_edge[0];
    float s = 1.f / (1.f + expf(-z));              // sigmoid
    ew[e] = s;
    atomicAdd(&deg[row[e]], s);                    // row-sum of adjacency
}

__global__ void k_dis(const float* __restrict__ deg, float* __restrict__ dis) {
    int i = blockIdx.x * blockDim.x + threadIdx.x;
    if (i >= GN) return;
    float d = deg[i];
    dis[i] = (d > 0.f) ? rsqrtf(d) : 0.f;          // D^-1/2 (deg >= 1 here)
}

__global__ void k_norm_edges(const float* __restrict__ ew,
                             const int* __restrict__ row,
                             const int* __restrict__ col,
                             const float* __restrict__ dis,
                             float* __restrict__ wn) {
    int e = blockIdx.x * blockDim.x + threadIdx.x;
    if (e >= GE) return;
    wn[e] = dis[row[e]] * ew[e] * dis[col[e]];
}

// weight padding: w1 (15x32) -> w1p (32x32, zero rows 15..31)
__global__ void k_pad_w1(const float* __restrict__ w1, float* __restrict__ w1p) {
    int i = blockIdx.x * blockDim.x + threadIdx.x;
    if (i >= 32 * 32) return;
    int k = i >> 5, n = i & 31;
    w1p[i] = (k < 15) ? w1[k * 32 + n] : 0.f;
}

// weight padding: wf3 (256x11) -> wf3p (256x16); bf3 (11) -> bf3p (16)
__global__ void k_pad_wf3(const float* __restrict__ wf3, const float* __restrict__ bf3,
                          float* __restrict__ wf3p, float* __restrict__ bf3p) {
    int i = blockIdx.x * blockDim.x + threadIdx.x;
    if (i >= 256 * 16) return;
    int k = i >> 4, n = i & 15;
    wf3p[i] = (n < GNC) ? wf3[k * GNC + n] : 0.f;
    if (i < 16) bf3p[i] = (i < GNC) ? bf3[i] : 0.f;
}

// t[i,0:F] = dis[i]^2 * h[i,0:F]   (self-loop term; ldt may exceed F)
__global__ void k_spmm_self(float* __restrict__ t, const float* __restrict__ h,
                            const float* __restrict__ dis,
                            int logF, int ldt, int total) {
    int i = blockIdx.x * blockDim.x + threadIdx.x;
    if (i >= total) return;
    int node = i >> logF;
    int f = i & ((1 << logF) - 1);
    float d = dis[node];
    t[node * ldt + f] = d * d * h[i];
}

// t[row,:] += wn[e] * h[col,:]   via f32 global atomics
__global__ void k_spmm_edge(float* __restrict__ t, const float* __restrict__ h,
                            const float* __restrict__ wn,
                            const int* __restrict__ row,
                            const int* __restrict__ col,
                            int logF, int ldt, int total) {
    int i = blockIdx.x * blockDim.x + threadIdx.x;
    if (i >= total) return;
    int e = i >> logF;
    int f = i & ((1 << logF) - 1);
    atomicAdd(&t[row[e] * ldt + f], wn[e] * h[(col[e] << logF) + f]);
}

// ---------------------------- WMMA GEMM ------------------------------------
// D(MxNc) = A(MxK) * B(KxNc) + bias  with v_wmma_f32_16x16x32_f16.
// REQUIRES: M%16==0, K%32==0, Nc%16==0 (caller pads); 16B-aligned pointers,
// lda/ldb/ldd multiples of 4. One wave owns one 16x16 output tile; fragment
// loads: A contiguous float4 (global_load_b128), B strided scalar; converted to f16 in VGPRs.

__global__ void k_wmma_gemm_bias(float* __restrict__ D,
                                 const float* __restrict__ A,
                                 const float* __restrict__ Bm,
                                 const float* __restrict__ bias,
                                 int M, int Nc, int K,
                                 int lda, int ldb, int ldd) {
    int wave = (blockIdx.x * blockDim.x + threadIdx.x) >> 5;
    int lane = threadIdx.x & 31;
    int tilesM = M >> 4;
    int tilesN = Nc >> 4;
    if (wave >= tilesM * tilesN) return;          // uniform per-wave exit
    int m0 = (wave % tilesM) << 4;
    int n0 = (wave / tilesM) << 4;
    int l15 = lane & 15, lhi = lane >> 4;

    // C/D layout: n = lane&15 (all vgprs), m = vgpr + 8*(lane>=16)
    int n = n0 + l15;
    float bv = bias[n];
    v8f c;
#pragma unroll
    for (int v = 0; v < 8; ++v) c[v] = bv;        // fold bias into accumulator

    // A frag: lane holds row m0+(lane&15); K-halves 0..7/16..23 (lanes<16)
    // or 8..15/24..31 (lanes>=16).  B frag: lane holds col n0+(lane&15), same K order.
    const float* ap = A + (m0 + l15) * lda + (lhi << 3);
    const float* bp = Bm + (lhi << 3) * ldb + n0 + l15;
    int bstep = ldb << 5;

    for (int k0 = 0; k0 < K; k0 += 32) {
        float4 a0 = *(const float4*)(ap);
        float4 a1 = *(const float4*)(ap + 4);
        float4 a2 = *(const float4*)(ap + 16);
        float4 a3 = *(const float4*)(ap + 20);
        float4 b0 = make_float4(bp[0], bp[ldb], bp[2 * ldb], bp[3 * ldb]);
        float4 b1 = make_float4(bp[4 * ldb], bp[5 * ldb], bp[6 * ldb], bp[7 * ldb]);
        float4 b2 = make_float4(bp[16 * ldb], bp[17 * ldb], bp[18 * ldb], bp[19 * ldb]);
        float4 b3 = make_float4(bp[20 * ldb], bp[21 * ldb], bp[22 * ldb], bp[23 * ldb]);
        v16h a, b;
        a[0]  = (_Float16)a0.x; a[1]  = (_Float16)a0.y;
        a[2]  = (_Float16)a0.z; a[3]  = (_Float16)a0.w;
        a[4]  = (_Float16)a1.x; a[5]  = (_Float16)a1.y;
        a[6]  = (_Float16)a1.z; a[7]  = (_Float16)a1.w;
        a[8]  = (_Float16)a2.x; a[9]  = (_Float16)a2.y;
        a[10] = (_Float16)a2.z; a[11] = (_Float16)a2.w;
        a[12] = (_Float16)a3.x; a[13] = (_Float16)a3.y;
        a[14] = (_Float16)a3.z; a[15] = (_Float16)a3.w;
        b[0]  = (_Float16)b0.x; b[1]  = (_Float16)b0.y;
        b[2]  = (_Float16)b0.z; b[3]  = (_Float16)b0.w;
        b[4]  = (_Float16)b1.x; b[5]  = (_Float16)b1.y;
        b[6]  = (_Float16)b1.z; b[7]  = (_Float16)b1.w;
        b[8]  = (_Float16)b2.x; b[9]  = (_Float16)b2.y;
        b[10] = (_Float16)b2.z; b[11] = (_Float16)b2.w;
        b[12] = (_Float16)b3.x; b[13] = (_Float16)b3.y;
        b[14] = (_Float16)b3.z; b[15] = (_Float16)b3.w;
        c = __builtin_amdgcn_wmma_f32_16x16x32_f16(
                false, a, false, b, (short)0, c, false, false);
        ap += 32;
        bp += bstep;
    }

    int mo = lhi << 3;
    float* dp = D + (m0 + mo) * ldd + n;
#pragma unroll
    for (int v = 0; v < 8; ++v) dp[v * ldd] = c[v];
}

// ---------------------------- BatchNorm ------------------------------------

__global__ void k_bn_stats(const float* __restrict__ Z, float* __restrict__ mu,
                           float* __restrict__ var, int M, int ld) {
    int colIdx = blockIdx.x;
    __shared__ float ss[256], ss2[256];
    float s = 0.f, s2 = 0.f;
    for (int r = threadIdx.x; r < M; r += blockDim.x) {
        float v = Z[r * ld + colIdx];
        s += v; s2 += v * v;
    }
    ss[threadIdx.x] = s; ss2[threadIdx.x] = s2;
    __syncthreads();
    for (int off = blockDim.x >> 1; off > 0; off >>= 1) {
        if (threadIdx.x < off) {
            ss[threadIdx.x]  += ss[threadIdx.x + off];
            ss2[threadIdx.x] += ss2[threadIdx.x + off];
        }
        __syncthreads();
    }
    if (threadIdx.x == 0) {
        float m = ss[0] / (float)M;
        float v = ss2[0] / (float)M - m * m;       // biased var == jnp.var
        mu[colIdx] = m;
        var[colIdx] = v > 0.f ? v : 0.f;
    }
}

__global__ void k_bn_relu(float* __restrict__ H, const float* __restrict__ Z,
                          const float* __restrict__ mu, const float* __restrict__ var,
                          const float* __restrict__ g, const float* __restrict__ be,
                          int cmask, int total) {
    int i = blockIdx.x * blockDim.x + threadIdx.x;
    if (i >= total) return;
    int c = i & cmask;                             // Nc is a power of two
    float v = g[c] * (Z[i] - mu[c]) * rsqrtf(var[c] + BN_EPS) + be[c];
    H[i] = v > 0.f ? v : 0.f;
}

// Head BN (batch of 16 rows): one thread per column does stats + normalize.
__global__ void k_head_bn_relu(float* __restrict__ O, const float* __restrict__ Z,
                               const float* __restrict__ g, const float* __restrict__ be,
                               int cols) {
    int c = blockIdx.x * blockDim.x + threadIdx.x;
    if (c >= cols) return;
    float s = 0.f, s2 = 0.f;
    for (int r = 0; r < GB; ++r) { float v = Z[r * cols + c]; s += v; s2 += v * v; }
    float m = s / (float)GB;
    float va = s2 / (float)GB - m * m; va = va > 0.f ? va : 0.f;
    float inv = rsqrtf(va + BN_EPS);
    for (int r = 0; r < GB; ++r) {
        float v = g[c] * (Z[r * cols + c] - m) * inv + be[c];
        O[r * cols + c] = v > 0.f ? v : 0.f;
    }
}

// ---------------------------- pooling + softmax ----------------------------

__global__ void k_zero(float* __restrict__ p, int n) {
    int i = blockIdx.x * blockDim.x + threadIdx.x;
    if (i < n) p[i] = 0.f;
}

// segment max over batch ids; h >= 0 (post-ReLU) so uint-ordered atomicMax is
// exact, and the 0-init matches the reference's empty-graph -> 0 semantics.
__global__ void k_pool_max(const float* __restrict__ h,
                           const int* __restrict__ batch,
                           float* __restrict__ emb) {
    int i = blockIdx.x * blockDim.x + threadIdx.x;
    if (i >= GN * 64) return;
    int node = i >> 6, f = i & 63;
    atomicMax((unsigned int*)&emb[batch[node] * 64 + f], __float_as_uint(h[i]));
}

__global__ void k_log_softmax(const float* __restrict__ logits,  // ld = 16
                              float* __restrict__ out) {
    int r = threadIdx.x;
    if (r >= GB) return;
    float mx = -1e30f;
    for (int j = 0; j < GNC; ++j) mx = fmaxf(mx, logits[r * 16 + j]);
    float s = 0.f;
    for (int j = 0; j < GNC; ++j) s += expf(logits[r * 16 + j] - mx);
    float ls = logf(s);
    for (int j = 0; j < GNC; ++j) out[r * GNC + j] = logits[r * 16 + j] - mx - ls;
}

// ---------------------------- host driver ----------------------------------

static inline void launch_gemm(float* D, const float* A, const float* Bm,
                               const float* bias, int M, int Nc, int K,
                               int lda, int ldb, int ldd, hipStream_t s) {
    int tiles = (M >> 4) * (Nc >> 4);
    int threads = tiles * 32;
    int grid = (threads + 127) / 128;
    k_wmma_gemm_bias<<<grid, 128, 0, s>>>(D, A, Bm, bias, M, Nc, K, lda, ldb, ldd);
}

extern "C" void kernel_launch(void* const* d_in, const int* in_sizes, int n_in,
                              void* d_out, int out_size, void* d_ws, size_t ws_size,
                              hipStream_t stream) {
    const float* x       = (const float*)d_in[0];
    const float* ea      = (const float*)d_in[1];
    const float* w_edge  = (const float*)d_in[2];
    const float* b_edge  = (const float*)d_in[3];
    const float* w1 = (const float*)d_in[4],  *b1 = (const float*)d_in[5];
    const float* g1 = (const float*)d_in[6],  *be1 = (const float*)d_in[7];
    const float* w2 = (const float*)d_in[8],  *b2 = (const float*)d_in[9];
    const float* g2 = (const float*)d_in[10], *be2 = (const float*)d_in[11];
    const float* w3 = (const float*)d_in[12], *b3 = (const float*)d_in[13];
    const float* g3 = (const float*)d_in[14], *be3 = (const float*)d_in[15];
    const float* wf1 = (const float*)d_in[16], *bf1 = (const float*)d_in[17];
    const float* gf1 = (const float*)d_in[18], *bef1 = (const float*)d_in[19];
    const float* wf2 = (const float*)d_in[20], *bf2 = (const float*)d_in[21];
    const float* gf2 = (const float*)d_in[22], *bef2 = (const float*)d_in[23];
    const float* wf3 = (const float*)d_in[24], *bf3 = (const float*)d_in[25];
    const int* edge_index = (const int*)d_in[26];
    const int* batch      = (const int*)d_in[27];
    const int* erow = edge_index;        // edge_index[0, :]
    const int* ecol = edge_index + GE;   // edge_index[1, :]

    float* out = (float*)d_out;          // [16*11 logprobs][16*64 emb]
    float* emb = out + GB * GNC;

    // workspace carve-up (floats)
    float* ws   = (float*)d_ws;
    float* ew   = ws;                 // E
    float* wn   = ew + GE;            // E
    float* deg  = wn + GE;            // N
    float* dis  = deg + GN;           // N
    float* xpad = dis + GN;           // N*16
    float* t    = xpad + GN * 16;     // N*64
    float* z    = t + GN * 64;        // N*64
    float* h    = z + GN * 64;        // N*64
    float* mu   = h + GN * 64;        // 512
    float* var  = mu + 512;           // 512
    float* o1r  = var + 512;          // 16*512
    float* o1   = o1r + GB * 512;     // 16*512
    float* o2r  = o1 + GB * 512;      // 16*256
    float* o2   = o2r + GB * 256;     // 16*256
    float* lg   = o2 + GB * 256;      // 16*16 (padded logits)
    float* w1p  = lg + GB * 16;       // 32*32 padded w1
    float* wf3p = w1p + 32 * 32;      // 256*16 padded wf3
    float* bf3p = wf3p + 256 * 16;    // 16 padded bf3

    const int TB = 256;
    // --- graph normalization + weight padding ---
    k_init_xpad_deg<<<(GN * 16 + TB - 1) / TB, TB, 0, stream>>>(x, xpad, deg);
    k_pad_w1<<<(32 * 32 + TB - 1) / TB, TB, 0, stream>>>(w1, w1p);
    k_pad_wf3<<<(256 * 16 + TB - 1) / TB, TB, 0, stream>>>(wf3, bf3, wf3p, bf3p);
    k_edge_weight_deg<<<(GE + TB - 1) / TB, TB, 0, stream>>>(ea, w_edge, b_edge, erow, ew, deg);
    k_dis<<<(GN + TB - 1) / TB, TB, 0, stream>>>(deg, dis);
    k_norm_edges<<<(GE + TB - 1) / TB, TB, 0, stream>>>(ew, erow, ecol, dis, wn);

    // --- GCN layer 1: F_in = 16 (padded, 15 real), t stored N x 32 ---
    k_zero<<<(GN * 32 + TB - 1) / TB, TB, 0, stream>>>(t, GN * 32);
    k_spmm_self<<<(GN * 16 + TB - 1) / TB, TB, 0, stream>>>(t, xpad, dis, 4, 32, GN * 16);
    k_spmm_edge<<<(GE * 16 + TB - 1) / TB, TB, 0, stream>>>(t, xpad, wn, erow, ecol, 4, 32, GE * 16);
    launch_gemm(z, t, w1p, b1, GN, 32, 32, 32, 32, 32, stream);
    k_bn_stats<<<32, 256, 0, stream>>>(z, mu, var, GN, 32);
    k_bn_relu<<<(GN * 32 + TB - 1) / TB, TB, 0, stream>>>(h, z, mu, var, g1, be1, 31, GN * 32);

    // --- GCN layer 2: 32 -> 64 ---
    k_spmm_self<<<(GN * 32 + TB - 1) / TB, TB, 0, stream>>>(t, h, dis, 5, 32, GN * 32);
    k_spmm_edge<<<(GE * 32 + TB - 1) / TB, TB, 0, stream>>>(t, h, wn, erow, ecol, 5, 32, GE * 32);
    launch_gemm(z, t, w2, b2, GN, 64, 32, 32, 64, 64, stream);
    k_bn_stats<<<64, 256, 0, stream>>>(z, mu, var, GN, 64);
    k_bn_relu<<<(GN * 64 + TB - 1) / TB, TB, 0, stream>>>(h, z, mu, var, g2, be2, 63, GN * 64);

    // --- GCN layer 3: 64 -> 64 ---
    k_spmm_self<<<(GN * 64 + TB - 1) / TB, TB, 0, stream>>>(t, h, dis, 6, 64, GN * 64);
    k_spmm_edge<<<(GE * 64 + TB - 1) / TB, TB, 0, stream>>>(t, h, wn, erow, ecol, 6, 64, GE * 64);
    launch_gemm(z, t, w3, b3, GN, 64, 64, 64, 64, 64, stream);
    k_bn_stats<<<64, 256, 0, stream>>>(z, mu, var, GN, 64);
    k_bn_relu<<<(GN * 64 + TB - 1) / TB, TB, 0, stream>>>(h, z, mu, var, g3, be3, 63, GN * 64);

    // --- global max pool -> emb (written straight into d_out) ---
    k_zero<<<(GB * 64 + TB - 1) / TB, TB, 0, stream>>>(emb, GB * 64);
    k_pool_max<<<(GN * 64 + TB - 1) / TB, TB, 0, stream>>>(h, batch, emb);

    // --- MLP head ---
    launch_gemm(o1r, emb, wf1, bf1, GB, 512, 64, 64, 512, 512, stream);
    k_head_bn_relu<<<(512 + TB - 1) / TB, TB, 0, stream>>>(o1, o1r, gf1, bef1, 512);
    launch_gemm(o2r, o1, wf2, bf2, GB, 256, 512, 512, 256, 256, stream);
    k_head_bn_relu<<<(256 + TB - 1) / TB, TB, 0, stream>>>(o2, o2r, gf2, bef2, 256);
    launch_gemm(lg, o2, wf3p, bf3p, GB, 16, 256, 256, 16, 16, stream);
    k_log_softmax<<<1, 32, 0, stream>>>(lg, out);
}
